// LlamaAttention_49057116454956
// MI455X (gfx1250) — matrix operand order repaired and checked by reading, not compiled, not run on hardware
//
#include <hip/hip_runtime.h>
#include <hip/hip_bf16.h>

// ---------------------------------------------------------------------------
// Problem constants (match reference): B=2, S=2048, D=2048, H=16, HD=128
// ---------------------------------------------------------------------------
#define B_ 2
#define S_ 2048
#define D_ 2048
#define H_ 16
#define HD_ 128
#define MN_ (B_ * S_)            // 4096 total rows
#define KC_ 64                   // K-chunk staged per async round
#define NCH_ (D_ / KC_)          // 32 chunks
#define SCALE_ 0.08838834764831845f   // 1/sqrt(128)
#define LOG_THETA_OVER_HALF 0.14391156250f  // ln(10000)/64

typedef __bf16 bf16_t;
typedef __attribute__((ext_vector_type(16))) __bf16 v16bf;
typedef __attribute__((ext_vector_type(8)))  __bf16 v8bf;
typedef __attribute__((ext_vector_type(8)))  float  v8f;
typedef int v4i __attribute__((vector_size(16)));

typedef __attribute__((address_space(1))) v4i gas_v4i;   // global
typedef __attribute__((address_space(3))) v4i las_v4i;   // LDS
typedef __attribute__((address_space(3))) bf16_t lds_bf16;

__device__ __forceinline__ bf16_t to_bf16(float f) {
  unsigned u = __builtin_bit_cast(unsigned, f);
  u += 0x7FFFu + ((u >> 16) & 1u);          // round-to-nearest-even
  unsigned short h = (unsigned short)(u >> 16);
  return __builtin_bit_cast(bf16_t, h);
}

__device__ __forceinline__ v8f wmma_bf16(v16bf a, v16bf b, v8f c) {
  return __builtin_amdgcn_wmma_f32_16x16x32_bf16(
      /*neg_a=*/false, a, /*neg_b=*/false, b,
      /*c_mod=*/(short)0, c, /*reuse_a=*/false, /*reuse_b=*/false);
}

// ------------------- CDNA5 async global->LDS copy (16 bytes) ----------------
__device__ __forceinline__ void async_ld16(const bf16_t* g, bf16_t* l) {
#if __has_builtin(__builtin_amdgcn_global_load_async_to_lds_b128)
  __builtin_amdgcn_global_load_async_to_lds_b128(
      (gas_v4i*)(v4i*)g, (las_v4i*)(v4i*)l, /*offset=*/0, /*cpol=*/0);
#else
  unsigned loff = (unsigned)(unsigned long long)(lds_bf16*)l;   // LDS byte offset
  unsigned long long ga = (unsigned long long)g;
  asm volatile("global_load_async_to_lds_b128 %0, %1, off"
               :: "v"(loff), "v"(ga) : "memory");
#endif
}
__device__ __forceinline__ void wait_async_all() {
  asm volatile("s_wait_asynccnt 0x0" ::: "memory");
}

// A fragment (16x32 bf16, row-major source, stride ld elements).
// Lane layout: row = lane&15; elems 0..15 -> K = k0+16*hs+i
// (hs = lane>>4), matching the contiguous B fragments.
__device__ __forceinline__ v16bf load_a_frag(const bf16_t* __restrict__ base,
                                             int ld, int row, int k0, int lane) {
  const int hs = lane >> 4;
  const bf16_t* p = base + (size_t)row * ld + k0 + 16 * hs;
  const v8bf lo = *(const v8bf*)(p);
  const v8bf hi = *(const v8bf*)(p + 8);
  v16bf a;
#pragma unroll
  for (int i = 0; i < 8; ++i) { a[i] = lo[i]; a[i + 8] = hi[i]; }
  return a;
}

// B fragment (32x16) from N-major operand in global memory (one 32B load).
__device__ __forceinline__ v16bf load_b_frag(const bf16_t* __restrict__ baseT,
                                             int ld, int nrow, int k0, int lane) {
  const int hs = lane >> 4;
  return *(const v16bf*)(baseT + (size_t)nrow * ld + k0 + 16 * hs);
}

// B fragment from an LDS-staged tile: rows of KC_ bf16, contiguous.
// ptr = tile + nrow*KC_ + kin + 16*hs  (32B aligned) -> two ds_load_b128.
__device__ __forceinline__ v16bf load_b_frag_lds(const bf16_t* tile,
                                                 int nrow, int kin, int lane) {
  const int hs = lane >> 4;
  return *(const v16bf*)(tile + nrow * KC_ + kin + 16 * hs);
}

// ---------------------------------------------------------------------------
// 1) x (f32) -> xb (bf16), 4 elems/thread
// ---------------------------------------------------------------------------
__global__ void cvt_x_kernel(const float* __restrict__ x, bf16_t* __restrict__ xb) {
  const int i = (blockIdx.x * blockDim.x + threadIdx.x) * 4;
  const float4 f = *(const float4*)(x + i);
  xb[i + 0] = to_bf16(f.x);
  xb[i + 1] = to_bf16(f.y);
  xb[i + 2] = to_bf16(f.z);
  xb[i + 3] = to_bf16(f.w);
}

// ---------------------------------------------------------------------------
// 2) weight transpose + convert: wt[n*D + k] = bf16(w[k*D + n])
// ---------------------------------------------------------------------------
__global__ void transpose_w_kernel(const float* __restrict__ w, bf16_t* __restrict__ wt) {
  const int i = blockIdx.x * blockDim.x + threadIdx.x;   // i = n*D + k
  const int n = i >> 11;
  const int k = i & (D_ - 1);
  wt[i] = to_bf16(w[(size_t)k * D_ + n]);
}

// ---------------------------------------------------------------------------
// 3) Q/K projection GEMM + fused RoPE.
// Block = 8 waves, all on the SAME head column pair (n1, n1+64), each wave a
// different 16-row M tile (block covers 128 rows). The 32 weight columns are
// async-staged into LDS (double buffered, KC_=64 deep) and shared by all
// waves; A fragments stream from global per wave.
// grid: mblk (MN/128 = 32)  x  pair (64)
// ---------------------------------------------------------------------------
__global__ void proj_rope_kernel(const bf16_t* __restrict__ A,
                                 const bf16_t* __restrict__ Wt,
                                 bf16_t* __restrict__ Out) {
  __shared__ bf16_t bsm[2][32 * KC_];                    // 2 x 4KB
  const int tid  = threadIdx.x;
  const int lane = tid & 31;
  const int wv   = tid >> 5;
  const int pair = blockIdx.x & 63;
  const int mblk = blockIdx.x >> 6;
  const int h = pair >> 2, g = pair & 3;
  const int n1 = h * HD_ + g * 16;
  const int n2 = n1 + 64;
  const int m0 = (mblk * 8 + wv) * 16;
  const int col = lane & 15, hs = lane >> 4;

  // staging map: 32 rows x 64 elems, 16B (8 elems) per thread
  const int srow = tid >> 3;                             // 0..31
  const int soff = (tid & 7) * 8;                        // 0..56
  const int grow = (srow < 16) ? (n1 + srow) : (n2 + srow - 16);

  v8f c1 = {}; v8f c2 = {};

  // prologue: stage chunk 0
  async_ld16(Wt + (size_t)grow * D_ + soff, &bsm[0][srow * KC_ + soff]);

  for (int ch = 0; ch < NCH_; ++ch) {
    wait_async_all();
    __syncthreads();
    if (ch + 1 < NCH_)
      async_ld16(Wt + (size_t)grow * D_ + (ch + 1) * KC_ + soff,
                 &bsm[(ch + 1) & 1][srow * KC_ + soff]);
    const bf16_t* tile = bsm[ch & 1];
    const int k0 = ch * KC_;
#pragma unroll
    for (int kin = 0; kin < KC_; kin += 32) {
      const v16bf a  = load_a_frag(A, D_, m0 + col, k0 + kin, lane);
      const v16bf b1 = load_b_frag_lds(tile, col,      kin, lane);
      const v16bf b2 = load_b_frag_lds(tile, 16 + col, kin, lane);
      c1 = wmma_bf16(a, b1, c1);
      c2 = wmma_bf16(a, b2, c2);
    }
    __syncthreads();
  }

  // RoPE: out[d] = x[d]*cos - x[d+64]*sin ; out[d+64] = x[d+64]*cos + x[d]*sin
  const int dh = g * 16 + col;                            // 0..63
  const float invf = __expf(-(float)dh * LOG_THETA_OVER_HALF);
#pragma unroll
  for (int r = 0; r < 8; ++r) {
    const int row = m0 + r + 8 * hs;
    const int pos = row & (S_ - 1);
    float sn, cs;
    __sincosf((float)pos * invf, &sn, &cs);
    const float q1 = c1[r], q2 = c2[r];
    Out[(size_t)row * D_ + n1 + col] = to_bf16(q1 * cs - q2 * sn);
    Out[(size_t)row * D_ + n2 + col] = to_bf16(q2 * cs + q1 * sn);
  }
}

// ---------------------------------------------------------------------------
// Shared body for the two "16x64 per wave, async-staged 64-column B" GEMMs.
// MODE 0: C = f32 [MN,D] (output projection)
// MODE 1: write V^T bf16: Vt[((b*H+h)*HD + dh)*S + s]
// grid: mblk (32) x ngrp (32); block covers 128 rows x 64 cols.
// ---------------------------------------------------------------------------
template <int MODE>
__global__ void gemm64_kernel(const bf16_t* __restrict__ A,
                              const bf16_t* __restrict__ Wt,
                              void* __restrict__ Cout) {
  __shared__ bf16_t bsm[2][64 * KC_];                    // 2 x 8KB
  const int tid  = threadIdx.x;
  const int lane = tid & 31;
  const int wv   = tid >> 5;
  const int ngrp = blockIdx.x & 31;
  const int mblk = blockIdx.x >> 5;
  const int n0 = ngrp * 64;
  const int m0 = (mblk * 8 + wv) * 16;
  const int col = lane & 15, hs = lane >> 4;

  // staging map: 64 rows x 64 elems, 32B (16 elems) per thread
  const int srow = tid >> 2;                             // 0..63
  const int soff = (tid & 3) * 16;                       // 0..48

  v8f c[4];
#pragma unroll
  for (int j = 0; j < 4; ++j) c[j] = (v8f){};

  {
    const bf16_t* g = Wt + (size_t)(n0 + srow) * D_ + soff;
    async_ld16(g,     &bsm[0][srow * KC_ + soff]);
    async_ld16(g + 8, &bsm[0][srow * KC_ + soff + 8]);
  }

  for (int ch = 0; ch < NCH_; ++ch) {
    wait_async_all();
    __syncthreads();
    if (ch + 1 < NCH_) {
      const bf16_t* g = Wt + (size_t)(n0 + srow) * D_ + (ch + 1) * KC_ + soff;
      bf16_t* l = &bsm[(ch + 1) & 1][srow * KC_ + soff];
      async_ld16(g, l);
      async_ld16(g + 8, l + 8);
    }
    const bf16_t* tile = bsm[ch & 1];
    const int k0 = ch * KC_;
#pragma unroll
    for (int kin = 0; kin < KC_; kin += 32) {
      const v16bf a = load_a_frag(A, D_, m0 + col, k0 + kin, lane);
#pragma unroll
      for (int j = 0; j < 4; ++j) {
        const v16bf bf = load_b_frag_lds(tile, j * 16 + col, kin, lane);
        c[j] = wmma_bf16(a, bf, c[j]);
      }
    }
    __syncthreads();
  }

  if (MODE == 0) {
    float* C = (float*)Cout;
#pragma unroll
    for (int j = 0; j < 4; ++j)
#pragma unroll
      for (int r = 0; r < 8; ++r)
        C[(size_t)(m0 + r + 8 * hs) * D_ + n0 + j * 16 + col] = c[j][r];
  } else {
    bf16_t* Vt = (bf16_t*)Cout;
#pragma unroll
    for (int j = 0; j < 4; ++j) {
      const int gc = n0 + j * 16 + col;
      const int h  = gc >> 7;
      const int dh = gc & (HD_ - 1);
#pragma unroll
      for (int r = 0; r < 8; ++r) {
        const int row = m0 + r + 8 * hs;
        const int b   = row >> 11;           // /S_
        const int s   = row & (S_ - 1);
        Vt[(((size_t)b * H_ + h) * HD_ + dh) * S_ + s] = to_bf16(c[j][r]);
      }
    }
  }
}

// ---------------------------------------------------------------------------
// 5) Causal flash attention: one wave per (b, h, 16-query tile).
//    Streams key tiles in pairs (K=32 for the P*V WMMA), online softmax,
//    P staged through per-wave LDS to re-layout C -> A fragment.
// ---------------------------------------------------------------------------
__global__ void attn_kernel(const bf16_t* __restrict__ Qb,
                            const bf16_t* __restrict__ Kb,
                            const bf16_t* __restrict__ Vt,
                            bf16_t* __restrict__ Ob) {
  __shared__ bf16_t plds[8 * 16 * 32];                   // 1KB per wave, 8 waves
  const int lane  = threadIdx.x & 31;
  const int wslot = threadIdx.x >> 5;
  const int wid   = blockIdx.x * (blockDim.x >> 5) + wslot;
  const int qt = wid & 127;
  const int h  = (wid >> 7) & 15;
  const int b  = wid >> 11;
  const int col = lane & 15, hs = lane >> 4;
  const int m0 = qt * 16;

  const size_t qbase = (size_t)b * S_ * D_ + (size_t)h * HD_;
  const bf16_t* Qp = Qb + qbase;
  const bf16_t* Kp = Kb + qbase;
  const bf16_t* Vp = Vt + ((size_t)(b * H_ + h) * HD_) * S_;
  bf16_t* myp = plds + wslot * 512;

  v16bf qf[4];
#pragma unroll
  for (int ks = 0; ks < 4; ++ks)
    qf[ks] = load_a_frag(Qp, D_, m0 + col, ks * 32, lane);

  v8f acc[8];
#pragma unroll
  for (int n = 0; n < 8; ++n) acc[n] = (v8f){};
  float m[8], l[8];
#pragma unroll
  for (int r = 0; r < 8; ++r) { m[r] = -1e30f; l[r] = 0.0f; }

  const int npairs = (qt >> 1) + 1;                      // causal: skip masked pairs
  for (int p = 0; p < npairs; ++p) {
    const int kb0 = p * 32;
    // --- scores: two 16x16 tiles over HD=128 (4 k-steps each) ---
    v8f s1 = {}, s2 = {};
#pragma unroll
    for (int ks = 0; ks < 4; ++ks) {
      const v16bf kf1 = load_b_frag(Kp, D_, kb0 + col,      ks * 32, lane);
      const v16bf kf2 = load_b_frag(Kp, D_, kb0 + 16 + col, ks * 32, lane);
      s1 = wmma_bf16(qf[ks], kf1, s1);
      s2 = wmma_bf16(qf[ks], kf2, s2);
    }
    // --- online softmax update (row = r + 8*hs, half-wave reductions) ---
    float alpha[8];
#pragma unroll
    for (int r = 0; r < 8; ++r) {
      const int qrow = m0 + r + 8 * hs;
      float v1 = s1[r] * SCALE_ + ((kb0 + col)      > qrow ? -1e9f : 0.0f);
      float v2 = s2[r] * SCALE_ + ((kb0 + 16 + col) > qrow ? -1e9f : 0.0f);
      float vm = fmaxf(v1, v2);
      vm = fmaxf(vm, __shfl_xor(vm, 1, 16));
      vm = fmaxf(vm, __shfl_xor(vm, 2, 16));
      vm = fmaxf(vm, __shfl_xor(vm, 4, 16));
      vm = fmaxf(vm, __shfl_xor(vm, 8, 16));
      const float mnew = fmaxf(m[r], vm);
      const float al = __expf(m[r] - mnew);
      const float e1 = __expf(v1 - mnew);
      const float e2 = __expf(v2 - mnew);
      float rs = e1 + e2;
      rs += __shfl_xor(rs, 1, 16);
      rs += __shfl_xor(rs, 2, 16);
      rs += __shfl_xor(rs, 4, 16);
      rs += __shfl_xor(rs, 8, 16);
      l[r] = al * l[r] + rs;
      m[r] = mnew;
      alpha[r] = al;
      myp[(r + 8 * hs) * 32 + col]      = to_bf16(e1);
      myp[(r + 8 * hs) * 32 + 16 + col] = to_bf16(e2);
    }
#pragma unroll
    for (int n = 0; n < 8; ++n)
#pragma unroll
      for (int r = 0; r < 8; ++r) acc[n][r] *= alpha[r];

    asm volatile("s_wait_dscnt 0x0" ::: "memory");       // LDS store -> load
    // --- P A-fragment (16x32) from LDS ---
    v16bf pa;
    {
      const bf16_t* pp = myp + col * 32 + 16 * hs;
      const v8bf lo = *(const v8bf*)(pp);
      const v8bf hi = *(const v8bf*)(pp + 8);
#pragma unroll
      for (int i = 0; i < 8; ++i) { pa[i] = lo[i]; pa[i + 8] = hi[i]; }
    }
    // --- acc += P @ V  (8 output column tiles of HD) ---
#pragma unroll
    for (int n = 0; n < 8; ++n) {
      const v16bf vf = *(const v16bf*)(Vp + (size_t)(n * 16 + col) * S_ + kb0 + 16 * hs);
      acc[n] = wmma_bf16(pa, vf, acc[n]);
    }
  }

  // --- normalize and write pre-Wo activations as bf16 [B*S, D] ---
#pragma unroll
  for (int r = 0; r < 8; ++r) {
    const float inv = 1.0f / l[r];
    const int row = b * S_ + m0 + r + 8 * hs;
#pragma unroll
    for (int n = 0; n < 8; ++n)
      Ob[(size_t)row * D_ + h * HD_ + n * 16 + col] = to_bf16(acc[n][r] * inv);
  }
}

// ---------------------------------------------------------------------------
// Host launcher
// ---------------------------------------------------------------------------
extern "C" void kernel_launch(void* const* d_in, const int* in_sizes, int n_in,
                              void* d_out, int out_size, void* d_ws, size_t ws_size,
                              hipStream_t stream) {
  (void)in_sizes; (void)n_in; (void)out_size; (void)ws_size;
  const float* x  = (const float*)d_in[0];
  const float* wq = (const float*)d_in[1];
  const float* wk = (const float*)d_in[2];
  const float* wv = (const float*)d_in[3];
  const float* wo = (const float*)d_in[4];
  // d_in[5] = additive causal mask; causality is applied analytically.
  float* out = (float*)d_out;

  char* ws = (char*)d_ws;
  size_t off = 0;
  auto alloc = [&](size_t bytes) -> void* {
    void* p = ws + off;
    off = (off + bytes + 255) & ~(size_t)255;
    return p;
  };
  const size_t act_bytes = (size_t)MN_ * D_ * sizeof(bf16_t);   // 16 MB
  const size_t w_bytes   = (size_t)D_ * D_ * sizeof(bf16_t);    // 8 MB
  bf16_t* xb  = (bf16_t*)alloc(act_bytes);
  bf16_t* wqT = (bf16_t*)alloc(w_bytes);
  bf16_t* wkT = (bf16_t*)alloc(w_bytes);
  bf16_t* wvT = (bf16_t*)alloc(w_bytes);
  bf16_t* woT = (bf16_t*)alloc(w_bytes);
  bf16_t* Qb  = (bf16_t*)alloc(act_bytes);
  bf16_t* Kb  = (bf16_t*)alloc(act_bytes);
  bf16_t* Vt  = (bf16_t*)alloc(act_bytes);
  bf16_t* Ob  = (bf16_t*)alloc(act_bytes);   // total ~117 MB

  // 1) convert inputs
  cvt_x_kernel<<<(MN_ * D_) / (256 * 4), 256, 0, stream>>>(x, xb);
  transpose_w_kernel<<<(D_ * D_) / 256, 256, 0, stream>>>(wq, wqT);
  transpose_w_kernel<<<(D_ * D_) / 256, 256, 0, stream>>>(wk, wkT);
  transpose_w_kernel<<<(D_ * D_) / 256, 256, 0, stream>>>(wv, wvT);
  transpose_w_kernel<<<(D_ * D_) / 256, 256, 0, stream>>>(wo, woT);

  // 2) projections (+RoPE fused for Q,K; transpose fused for V)
  proj_rope_kernel<<<(MN_ / 128) * 64, 256, 0, stream>>>(xb, wqT, Qb);
  proj_rope_kernel<<<(MN_ / 128) * 64, 256, 0, stream>>>(xb, wkT, Kb);
  gemm64_kernel<1><<<(MN_ / 128) * (D_ / 64), 256, 0, stream>>>(xb, wvT, (void*)Vt);

  // 3) causal flash attention
  attn_kernel<<<(B_ * H_ * (S_ / 16)) / 8, 256, 0, stream>>>(Qb, Kb, Vt, Ob);

  // 4) output projection
  gemm64_kernel<0><<<(MN_ / 128) * (D_ / 64), 256, 0, stream>>>(Ob, woT, (void*)out);
}
